// Attention_43087111914327
// MI455X (gfx1250) — matrix operand order repaired and checked by reading, not compiled, not run on hardware
//
#include <hip/hip_runtime.h>
#include <hip/hip_bf16.h>

// MI455X (gfx1250) local-attention pipeline, all GEMMs via v_wmma_f32_16x16x32_bf16.
// B=4, N=2048 (16x128 grid), DIM=512, 8 heads x 64, local mask 7x11.
// Compute-bound (~25 GFLOP after window pruning, ~100MB traffic, L2-resident).

typedef __attribute__((ext_vector_type(16))) __bf16 v16bf;
typedef __attribute__((ext_vector_type(8)))  float  v8f;

#define DIMC   512
#define NHEADS 8
#define HDIMC  64
#define HH     16
#define WWID   128
#define NTOK   2048          // HH*WWID
#define BBAT   4
#define MTOT   (BBAT*NTOK)   // 8192

static __device__ __forceinline__ unsigned short f2bf(float f) {
  unsigned u = __builtin_bit_cast(unsigned, f);
  unsigned r = u + 0x7FFFu + ((u >> 16) & 1u);   // round-to-nearest-even
  return (unsigned short)(r >> 16);
}

union FragU { v16bf v; unsigned u[8]; };

// A operand, 16x32 bf16 (ISA 7.12.2): lane m = lane&15; VGPR j holds K pair p
// with p = j + 4*(j>>2) + 4*hi  (lo lanes: K0..7,16..23; hi lanes: K8..15,24..31)
static __device__ __forceinline__ v16bf load_fragA(const unsigned short* lds,
                                                   int row, int rowStrideBytes,
                                                   int kByteOff, int hi) {
  FragU f;
  const char* base = (const char*)lds + row * rowStrideBytes + kByteOff;
#pragma unroll
  for (int j = 0; j < 8; ++j) {
    int p = j + 4 * (j >> 2) + 4 * hi;
    f.u[j] = *(const unsigned*)(base + 4 * p);
  }
  return f.v;
}

// B operand, 32x16 bf16: lane n = lane&15; same K order as A
// (lo lanes: K0..7,16..23; hi lanes: K8..15,24..31).
static __device__ __forceinline__ v16bf load_fragB(const unsigned short* lds,
                                                   int row, int rowStrideBytes,
                                                   int kByteOff, int hi) {
  FragU f;
  const char* base = (const char*)lds + row * rowStrideBytes + kByteOff;
#pragma unroll
  for (int j = 0; j < 8; ++j) f.u[j] = *(const unsigned*)(base + 4 * (j + 4 * (j >> 2) + 4 * hi));
  return f.v;
}

static __device__ __forceinline__ v8f wmma_bf16(v16bf a, v16bf b, v8f c) {
  return __builtin_amdgcn_wmma_f32_16x16x32_bf16(false, a, false, b,
                                                 (short)0, c, false, false);
}

// ---------------------------------------------------------------- convert f32 -> bf16
__global__ __launch_bounds__(256) void cvt_kernel(const float* __restrict__ src,
                                                  unsigned short* __restrict__ dst,
                                                  int n) {
  int i = blockIdx.x * 256 + threadIdx.x;
  if (i < n) dst[i] = f2bf(src[i]);
}

// ---------------------------------------------------------------- WMMA GEMM
// C[8192 x Nn] = A[8192 x 512](bf16) * Bw[512 x Nn](bf16) + bias
// MODE 0: Nn=1536, scatter bf16 to q/k/v (q scaled by 0.125)
// MODE 1: Nn=512,  f32 out to outF
// Double-buffered LDS (2x16KB), global fetch for step i+1 issued before the
// WMMAs of step i; one barrier per K-step; prefetch two K-steps further ahead.
template <int MODE>
__global__ __launch_bounds__(256)
void gemm_kernel(const unsigned short* __restrict__ A,
                 const unsigned short* __restrict__ Bw,
                 const float* __restrict__ bias,
                 int Nn,
                 unsigned short* __restrict__ qb,
                 unsigned short* __restrict__ kbuf,
                 unsigned short* __restrict__ vbuf,
                 float* __restrict__ outF) {
  __shared__ __align__(16) unsigned short ldsA[2][128 * 32];   // [m][k]
  __shared__ __align__(16) unsigned short ldsB[2][128 * 32];   // [n][k] (transposed)

  const int tid  = threadIdx.x;
  const int wave = tid >> 5, lane = tid & 31;
  const int hi = lane >> 4, ln = lane & 15;
  const int waveM = wave & 3;   // 4 strips of 32 rows
  const int waveN = wave >> 2;  // 2 strips of 64 cols
  const int m0 = blockIdx.x * 128;
  const int n0 = blockIdx.y * 128;

  v8f acc[2][4];
#pragma unroll
  for (int tm = 0; tm < 2; ++tm)
#pragma unroll
    for (int tn = 0; tn < 4; ++tn)
#pragma unroll
      for (int c = 0; c < 8; ++c) acc[tm][tn][c] = 0.0f;

  unsigned short ra[16], rb[16];
  auto fetch = [&](int k0) {
    if (k0 + 64 < 512) {  // speculative prefetch two steps further ahead
      __builtin_prefetch(&A[(size_t)m0 * 512 + k0 + 64], 0, 0);
      __builtin_prefetch(&Bw[(size_t)(k0 + 64) * Nn + n0], 0, 0);
    }
#pragma unroll
    for (int it = 0; it < 16; ++it) {
      int i = tid + it * 256;
      int m = i >> 5, k = i & 31;
      ra[it] = A[(size_t)(m0 + m) * 512 + k0 + k];
      int kk = i >> 7, n = i & 127;
      rb[it] = Bw[(size_t)(k0 + kk) * Nn + n0 + n];
    }
  };
  auto stash = [&](int buf) {
#pragma unroll
    for (int it = 0; it < 16; ++it) {
      int i = tid + it * 256;
      int m = i >> 5, k = i & 31;
      ldsA[buf][m * 32 + k] = ra[it];
      int kk = i >> 7, n = i & 127;
      ldsB[buf][n * 32 + kk] = rb[it];
    }
  };

  fetch(0);
  stash(0);
  for (int i = 0; i < 16; ++i) {
    __syncthreads();                    // stash(i) visible; buf (i+1)&1 free
    if (i + 1 < 16) fetch((i + 1) * 32);
    const int cur = i & 1;
    v16bf bfr[4];
#pragma unroll
    for (int tn = 0; tn < 4; ++tn)
      bfr[tn] = load_fragB(ldsB[cur], waveN * 64 + tn * 16 + ln, 64, 0, hi);
#pragma unroll
    for (int tm = 0; tm < 2; ++tm) {
      v16bf afr = load_fragA(ldsA[cur], waveM * 32 + tm * 16 + ln, 64, 0, hi);
#pragma unroll
      for (int tn = 0; tn < 4; ++tn)
        acc[tm][tn] = wmma_bf16(afr, bfr[tn], acc[tm][tn]);
    }
    if (i + 1 < 16) stash((i + 1) & 1);
  }

  // epilogue
#pragma unroll
  for (int tm = 0; tm < 2; ++tm) {
#pragma unroll
    for (int tn = 0; tn < 4; ++tn) {
      int nl = n0 + waveN * 64 + tn * 16 + ln;
      float bv = bias[nl];
#pragma unroll
      for (int r = 0; r < 8; ++r) {
        int ml = m0 + waveM * 32 + tm * 16 + r + 8 * hi;
        float v = acc[tm][tn][r] + bv;
        if constexpr (MODE == 0) {
          int sel  = nl >> 9;          // 0=q 1=k 2=v
          int hn   = nl & 511;
          int head = hn >> 6;
          int d    = hn & 63;
          int b    = ml >> 11;
          int tok  = ml & 2047;
          if (sel == 0) v *= 0.125f;   // head_dim^-0.5
          unsigned short* dst = (sel == 0) ? qb : (sel == 1) ? kbuf : vbuf;
          dst[(((size_t)(b * NHEADS + head)) * NTOK + tok) * HDIMC + d] = f2bf(v);
        } else {
          outF[(size_t)ml * 512 + nl] = v;
        }
      }
    }
  }
}

// ---------------------------------------------------------------- local attention
// grid = B*heads*H = 512 blocks of 256 threads (8 waves). Each block handles the
// 128 queries of one grid-row h for one (b, head). The h-window (hk=7) is the
// h2 loop; the w-band (|dw|<=5) is a register mask. Each wave processes only a
// 64-key window aligned to its 16-query strip (covers the 26-wide band exactly),
// halving WMMA work vs. the full 128-key row.
__global__ __launch_bounds__(256)
void attn_kernel(const unsigned short* __restrict__ qb,
                 const unsigned short* __restrict__ kb,
                 const unsigned short* __restrict__ vb,
                 unsigned short* __restrict__ ctx) {
  __shared__ __align__(16) unsigned short ksm[128 * 64];  // [key][d]
  __shared__ __align__(16) unsigned short vt[64 * 128];   // [d][key] (transposed)
  __shared__ __align__(16) unsigned short ps[128 * 64];   // [q][winkey] bf16 probs

  const int tid  = threadIdx.x;
  const int wave = tid >> 5, lane = tid & 31;
  const int hi = lane >> 4, ln = lane & 15;

  int bid  = blockIdx.x;
  int hq   = bid & 15;  bid >>= 4;
  int head = bid & 7;   bid >>= 3;
  int b    = bid;
  const size_t headBase = ((size_t)(b * NHEADS + head)) * NTOK * HDIMC;

  // per-wave 64-key window base (multiple of 16, covers [16w-5, 16w+20])
  int kwb = 16 * wave - 16;
  kwb = (kwb < 0) ? 0 : (kwb > 64) ? 64 : kwb;

  // Q fragments straight from global (L2-resident), reused for all key rows
  v16bf aqf[2];
  {
    const char* qrow = (const char*)(qb + headBase + (size_t)hq * WWID * HDIMC) +
                       (size_t)(wave * 16 + ln) * 128;
#pragma unroll
    for (int kstep = 0; kstep < 2; ++kstep) {
      FragU f;
#pragma unroll
      for (int j = 0; j < 8; ++j) {
        int p = j + 4 * (j >> 2) + 4 * hi;
        f.u[j] = *(const unsigned*)(qrow + kstep * 64 + 4 * p);
      }
      aqf[kstep] = f.v;
    }
  }

  float rowm[8], rowl[8];
#pragma unroll
  for (int r = 0; r < 8; ++r) { rowm[r] = -1e30f; rowl[r] = 0.0f; }
  v8f accO[4];
#pragma unroll
  for (int tn = 0; tn < 4; ++tn)
#pragma unroll
    for (int c = 0; c < 8; ++c) accO[tn][c] = 0.0f;

  const int h2lo = (hq - 3 < 0) ? 0 : hq - 3;
  const int h2hi = (hq + 3 > 15) ? 15 : hq + 3;

  for (int h2 = h2lo; h2 <= h2hi; ++h2) {
    __syncthreads();  // previous row's K/V reads done before overwrite
    const unsigned short* kg = kb + headBase + (size_t)h2 * WWID * HDIMC;
    const unsigned short* vg = vb + headBase + (size_t)h2 * WWID * HDIMC;
    {  // stage full key row [128 x 64]
      const uint2* src = (const uint2*)kg;
      uint2* dst = (uint2*)ksm;
#pragma unroll
      for (int it = 0; it < 8; ++it) dst[tid + it * 256] = src[tid + it * 256];
    }
#pragma unroll
    for (int it = 0; it < 32; ++it) {   // V transposed -> contiguous-K B operand
      int i = tid + it * 256;
      int key = i >> 6, d = i & 63;
      vt[d * 128 + key] = vg[i];
    }
    __syncthreads();

    // S = Q * K^T over this wave's 64-key window (K-dim 64 -> 2 steps)
    v8f sacc[4];
#pragma unroll
    for (int tn = 0; tn < 4; ++tn)
#pragma unroll
      for (int c = 0; c < 8; ++c) sacc[tn][c] = 0.0f;
#pragma unroll
    for (int kstep = 0; kstep < 2; ++kstep) {
#pragma unroll
      for (int tn = 0; tn < 4; ++tn) {
        v16bf bk = load_fragB(ksm, kwb + tn * 16 + ln, 128, kstep * 64, hi);
        sacc[tn] = wmma_bf16(aqf[kstep], bk, sacc[tn]);
      }
    }

    // banded mask + online softmax (row stats replicated across 16-lane half)
#pragma unroll
    for (int r = 0; r < 8; ++r) {
      int qw = wave * 16 + r + 8 * hi;
      float sv[4];
      float mblk = -1e30f;
#pragma unroll
      for (int tn = 0; tn < 4; ++tn) {
        int dw = (kwb + tn * 16 + ln) - qw;
        bool ok = (dw >= -5) && (dw <= 5);
        float s = ok ? sacc[tn][r] : -1e30f;
        sv[tn] = s;
        mblk = fmaxf(mblk, s);
      }
#pragma unroll
      for (int m = 1; m < 16; m <<= 1) mblk = fmaxf(mblk, __shfl_xor(mblk, m, 32));
      float newm = fmaxf(rowm[r], mblk);
      float corr = __expf(rowm[r] - newm);
      float lsum = 0.0f;
#pragma unroll
      for (int tn = 0; tn < 4; ++tn) {
        float p = (sv[tn] > -1e29f) ? __expf(sv[tn] - newm) : 0.0f;
        lsum += p;
        ps[(size_t)(wave * 16 + r + 8 * hi) * 64 + tn * 16 + ln] = f2bf(p);
      }
#pragma unroll
      for (int m = 1; m < 16; m <<= 1) lsum += __shfl_xor(lsum, m, 32);
      rowl[r] = rowl[r] * corr + lsum;
      rowm[r] = newm;
#pragma unroll
      for (int tn = 0; tn < 4; ++tn) accO[tn][r] *= corr;
    }

    // O += P * V  (each wave reads only its own 16 rows of ps -> no barrier)
#pragma unroll
    for (int kstep = 0; kstep < 2; ++kstep) {
      v16bf ap = load_fragA(ps, wave * 16 + ln, 128, kstep * 64, hi);
#pragma unroll
      for (int tn = 0; tn < 4; ++tn) {
        v16bf bv = load_fragB(vt, tn * 16 + ln, 256, kwb * 2 + kstep * 64, hi);
        accO[tn] = wmma_bf16(ap, bv, accO[tn]);
      }
    }
  }

  // normalize and write ctx [B, N, heads*64] (bf16) for the projection GEMM
  const size_t outBase = ((size_t)(b * NTOK + hq * WWID)) * DIMC + head * HDIMC;
#pragma unroll
  for (int r = 0; r < 8; ++r) {
    float inv = 1.0f / rowl[r];     // every query attends at least itself
    int qw = wave * 16 + r + 8 * hi;
#pragma unroll
    for (int tn = 0; tn < 4; ++tn) {
      int d = tn * 16 + ln;
      ctx[outBase + (size_t)qw * DIMC + d] = f2bf(accO[tn][r] * inv);
    }
  }
}

// ---------------------------------------------------------------- launch
extern "C" void kernel_launch(void* const* d_in, const int* in_sizes, int n_in,
                              void* d_out, int out_size, void* d_ws, size_t ws_size,
                              hipStream_t stream) {
  const float* x      = (const float*)d_in[0];
  const float* w_qkv  = (const float*)d_in[1];
  const float* b_qkv  = (const float*)d_in[2];
  const float* w_proj = (const float*)d_in[3];
  const float* b_proj = (const float*)d_in[4];
  float* out = (float*)d_out;

  char* ws = (char*)d_ws;
  size_t off = 0;
  auto carve = [&](size_t bytes) -> void* {
    void* p = ws + off;
    off = (off + bytes + 255) & ~(size_t)255;
    return p;
  };
  unsigned short* xb     = (unsigned short*)carve((size_t)MTOT * DIMC * 2);        // 8 MB
  unsigned short* wqkvb  = (unsigned short*)carve((size_t)DIMC * 3 * DIMC * 2);    // 1.5 MB
  unsigned short* wprojb = (unsigned short*)carve((size_t)DIMC * DIMC * 2);        // 0.5 MB
  unsigned short* qbuf   = (unsigned short*)carve((size_t)MTOT * DIMC * 2);        // 8 MB
  unsigned short* kbuf   = (unsigned short*)carve((size_t)MTOT * DIMC * 2);        // 8 MB
  unsigned short* vbuf   = (unsigned short*)carve((size_t)MTOT * DIMC * 2);        // 8 MB
  unsigned short* ctx    = (unsigned short*)carve((size_t)MTOT * DIMC * 2);        // 8 MB
  (void)ws_size; (void)in_sizes; (void)n_in; (void)out_size;

  const int nX  = MTOT * DIMC;           // 4194304
  const int nWq = DIMC * 3 * DIMC;       // 786432
  const int nWp = DIMC * DIMC;           // 262144
  cvt_kernel<<<(nX + 255) / 256, 256, 0, stream>>>(x, xb, nX);
  cvt_kernel<<<(nWq + 255) / 256, 256, 0, stream>>>(w_qkv, wqkvb, nWq);
  cvt_kernel<<<(nWp + 255) / 256, 256, 0, stream>>>(w_proj, wprojb, nWp);

  // QKV projection: [8192x512] x [512x1536]
  gemm_kernel<0><<<dim3(MTOT / 128, 1536 / 128), 256, 0, stream>>>(
      xb, wqkvb, b_qkv, 1536, qbuf, kbuf, vbuf, nullptr);

  // Local attention: one block per (b, head, grid-row)
  attn_kernel<<<BBAT * NHEADS * HH, 256, 0, stream>>>(qbuf, kbuf, vbuf, ctx);

  // Output projection: [8192x512] x [512x512] -> f32 out
  gemm_kernel<1><<<dim3(MTOT / 128, 512 / 128), 256, 0, stream>>>(
      ctx, wprojb, b_proj, 512, nullptr, nullptr, nullptr, out);
}
